// MaxMPNN_47132971107062
// MI455X (gfx1250) — matrix operand order repaired and checked by reading, not compiled, not run on hardware
//
#include <hip/hip_runtime.h>

// ---------------------------------------------------------------------------
// MaxMPNN on MI455X (gfx1250, wave32, WMMA).
//   GEMM1: Xbf(G,bf16) [20000 x 1440] @ W1ᵀ -> C1 [20000 x 2880] fp32 (A ‖ Bn)
//   top2 : per-channel top-2 of Bn over nodes
//   build: max_msg = A + bm + self-excluded max  -> bf16 into Xbf[:,1440:2880]
//   GEMM2: Xbf [20000 x 2880] @ W2ᵀ + bu -> out [20000 x 1433] fp32
// Matmuls run on v_wmma_f32_16x16x32_bf16 (fp32 accum), double-buffered LDS,
// with GLOBAL_LOAD_ASYNC_TO_LDS_B128 (ASYNCcnt) for full interior tiles.
// ---------------------------------------------------------------------------

typedef __attribute__((ext_vector_type(16))) __bf16        v16bf;
typedef __attribute__((ext_vector_type(8)))  float         v8f;
typedef __attribute__((ext_vector_type(4)))  unsigned int  u32x4;

union FragCast { u32x4 u[2]; v16bf v; };

constexpr int NNODES = 20000;
constexpr int D      = 1433;
constexpr int DP     = 1440;   // channels padded to multiple of 32 (WMMA K)
constexpr int K2     = 2880;   // 2 * DP (concatenated [G, max_msg])

#if __has_builtin(__builtin_amdgcn_s_wait_asynccnt)
#define WAIT_ASYNC(n) __builtin_amdgcn_s_wait_asynccnt(n)
#else
#define WAIT_ASYNC(n) asm volatile("s_wait_asynccnt %0" ::"i"(n) : "memory")
#endif

// ---------------- prep: fp32 -> bf16 conversions (zero-padded) -------------

__global__ void k_convert_g(const float* __restrict__ G, __bf16* __restrict__ X) {
    size_t idx = (size_t)blockIdx.x * blockDim.x + threadIdx.x;
    if (idx >= (size_t)NNODES * DP) return;
    int n = (int)(idx / DP), k = (int)(idx % DP);
    float v = (k < D) ? G[(size_t)n * D + k] : 0.f;
    X[(size_t)n * K2 + k] = (__bf16)v;
}

// W1[j][k] (j = output col of GEMM1): j<D -> Wm[j, k] (self part A),
// D<=j<2D -> Wm[j-D, D+k] (neighbor part Bn). Stored [2880][1440] bf16.
__global__ void k_convert_w1(const float* __restrict__ Wm, __bf16* __restrict__ W1) {
    size_t idx = (size_t)blockIdx.x * blockDim.x + threadIdx.x;
    if (idx >= (size_t)K2 * DP) return;
    int j = (int)(idx / DP), k = (int)(idx % DP);
    float v = 0.f;
    if (k < D) {
        if (j < D)          v = Wm[(size_t)j * (2 * D) + k];
        else if (j < 2 * D) v = Wm[(size_t)(j - D) * (2 * D) + D + k];
    }
    W1[idx] = (__bf16)v;
}

// W2[e][k] (e = output col of GEMM2): k<1440 -> Wu[e, k] (G part),
// k>=1440 -> Wu[e, D + (k-1440)] (max_msg part). Stored [1440][2880] bf16.
__global__ void k_convert_w2(const float* __restrict__ Wu, __bf16* __restrict__ W2) {
    size_t idx = (size_t)blockIdx.x * blockDim.x + threadIdx.x;
    if (idx >= (size_t)DP * K2) return;
    int e = (int)(idx / K2), k = (int)(idx % K2);
    float v = 0.f;
    if (e < D) {
        if (k < D)                      v = Wu[(size_t)e * (2 * D) + k];
        else if (k >= DP && k < DP + D) v = Wu[(size_t)e * (2 * D) + D + (k - DP)];
    }
    W2[idx] = (__bf16)v;
}

// ---------------- per-channel top-2 over nodes (coalesced) ------------------

__global__ void k_top2(const float* __restrict__ C1,
                       float* __restrict__ m1o, float* __restrict__ m2o,
                       int* __restrict__ a1o) {
    int ch = blockIdx.x * blockDim.x + threadIdx.x;
    if (ch >= D) return;
    float m1 = -3.402823466e38f, m2 = -3.402823466e38f;
    int a1 = -1;
    const float* p = C1 + D + ch;            // Bn column
    for (int n = 0; n < NNODES; ++n) {
        float v = p[(size_t)n * K2];
        if (v > m1)      { m2 = m1; m1 = v; a1 = n; }
        else if (v > m2) { m2 = v; }
    }
    m1o[ch] = m1; m2o[ch] = m2; a1o[ch] = a1;
}

// max_msg = A + bm + (argmax==self ? max2 : max1), write bf16 into Xbf[:,DP:]
__global__ void k_build_msg(const float* __restrict__ C1, const float* __restrict__ bm,
                            const float* __restrict__ m1, const float* __restrict__ m2,
                            const int* __restrict__ a1, __bf16* __restrict__ X) {
    size_t idx = (size_t)blockIdx.x * blockDim.x + threadIdx.x;
    if (idx >= (size_t)NNODES * DP) return;
    int n = (int)(idx / DP), k = (int)(idx % DP);
    float v = 0.f;
    if (k < D) {
        float mv = (a1[k] == n) ? m2[k] : m1[k];
        v = C1[(size_t)n * K2 + k] + bm[k] + mv;
    }
    X[(size_t)n * K2 + DP + k] = (__bf16)v;
}

// ---------------- WMMA GEMM: C[m][j] = sum_k A[m][k] * B[j][k] (+bias) ------
// Block tile 128x128, 8 waves, each wave 4(M) x 2(N) 16x16 fragments, K step
// 32, double-buffered LDS; async-to-LDS fast path for interior tiles.

__global__ __launch_bounds__(256, 2)
void k_wmma_gemm(const __bf16* __restrict__ A, int lda,
                 const __bf16* __restrict__ B, int ldb, int bRows,
                 float* __restrict__ C, int ldc, int mRows, int nValid,
                 int kIters, const float* __restrict__ bias)
{
    __shared__ __align__(16) __bf16 lA[2][128][40];  // +8 halves pad per row
    __shared__ __align__(16) __bf16 lB[2][128][40];
    constexpr unsigned LDSBUF = 128 * 40 * sizeof(__bf16);   // 10240 B

    const int tid  = threadIdx.x;
    const int lane = tid & 31;
    const int wave = tid >> 5;
    const int wm   = (wave >> 2) * 64;   // wave M base within tile (0,64)
    const int wn   = (wave & 3) * 32;    // wave N base within tile (0..96)
    const int lrow = lane & 15;
    const int lhi  = lane >> 4;          // 0 | 1 (upper half-wave)

    v8f acc[4][2];
    #pragma unroll
    for (int mi = 0; mi < 4; ++mi)
        #pragma unroll
        for (int ni = 0; ni < 2; ++ni)
            #pragma unroll
            for (int i = 0; i < 8; ++i) acc[mi][ni][i] = 0.f;

    // cooperative tile fill: each thread owns one 32B row-segment per matrix
    const int row = tid >> 1;            // 0..127
    const int seg = (tid & 1) * 16;      // halves
    const size_t gRowA = (size_t)blockIdx.y * 128 + row;
    const size_t gRowB = (size_t)blockIdx.x * 128 + row;
    const bool aOk = gRowA < (size_t)mRows;
    const bool bOk = gRowB < (size_t)bRows;
    const __bf16* aPtr = A + (aOk ? gRowA : 0) * (size_t)lda + seg;
    const __bf16* bPtr = B + (bOk ? gRowB : 0) * (size_t)ldb + seg;

    // frag-gather + 8x WMMA from one LDS buffer
    auto compute = [&](int b) {
        // B frag (32x16): lane holds col lrow, K = lhi*16 .. +15 contiguous
        FragCast bfrag[2];
        #pragma unroll
        for (int ni = 0; ni < 2; ++ni) {
            const __bf16* p = &lB[b][wn + ni * 16 + lrow][lhi * 16];
            bfrag[ni].u[0] = *(const u32x4*)(p);
            bfrag[ni].u[1] = *(const u32x4*)(p + 8);
        }
        // A frag (16x32): lane holds row lrow, K = lhi*16 .. +15 contiguous
        FragCast afrag[4];
        #pragma unroll
        for (int mi = 0; mi < 4; ++mi) {
            const __bf16* p = &lA[b][wm + mi * 16 + lrow][lhi * 16];
            afrag[mi].u[0] = *(const u32x4*)(p);
            afrag[mi].u[1] = *(const u32x4*)(p + 8);
        }
        #pragma unroll
        for (int mi = 0; mi < 4; ++mi)
            #pragma unroll
            for (int ni = 0; ni < 2; ++ni)
                acc[mi][ni] = __builtin_amdgcn_wmma_f32_16x16x32_bf16(
                    false, afrag[mi].v, false, bfrag[ni].v,
                    (short)0, acc[mi][ni], false, false);
    };

    const bool fullTile = ((size_t)(blockIdx.y + 1) * 128 <= (size_t)mRows) &&
                          ((size_t)(blockIdx.x + 1) * 128 <= (size_t)bRows);

    if (fullTile) {
        // -------- async fast path: GLOBAL_LOAD_ASYNC_TO_LDS_B128 ----------
        // IOFFSET is added to BOTH lds and global addrs, so one address pair
        // + offset:16 moves this thread's full 32B segment.
        const unsigned laBase = (unsigned)(size_t)&lA[0][row][seg];
        const unsigned lbBase = (unsigned)(size_t)&lB[0][row][seg];
        auto issue = [&](int buf) {
            unsigned la = laBase + (unsigned)buf * LDSBUF;
            unsigned lb = lbBase + (unsigned)buf * LDSBUF;
            asm volatile("global_load_async_to_lds_b128 %0, %1, off"
                         ::"v"(la), "v"(aPtr) : "memory");
            asm volatile("global_load_async_to_lds_b128 %0, %1, off offset:16"
                         ::"v"(la), "v"(aPtr) : "memory");
            asm volatile("global_load_async_to_lds_b128 %0, %1, off"
                         ::"v"(lb), "v"(bPtr) : "memory");
            asm volatile("global_load_async_to_lds_b128 %0, %1, off offset:16"
                         ::"v"(lb), "v"(bPtr) : "memory");
            aPtr += 32; bPtr += 32;
        };
        issue(0);
        for (int it = 0; it < kIters; ++it) {
            const int b = it & 1;
            if (it + 1 < kIters) {
                issue(1 - b);
                WAIT_ASYNC(4);   // the 4 just-issued may remain outstanding
            } else {
                WAIT_ASYNC(0);
            }
            __syncthreads();     // tile `b` visible to all waves
            compute(b);
        }
    } else {
        // -------- guarded sync path (edge tiles), double-buffered ---------
        u32x4 ra0, ra1, rb0, rb1;
        auto loadTile = [&]() {
            ra0 = u32x4{0,0,0,0}; ra1 = u32x4{0,0,0,0};
            rb0 = u32x4{0,0,0,0}; rb1 = u32x4{0,0,0,0};
            if (aOk) { ra0 = *(const u32x4*)(aPtr); ra1 = *(const u32x4*)(aPtr + 8); }
            if (bOk) { rb0 = *(const u32x4*)(bPtr); rb1 = *(const u32x4*)(bPtr + 8); }
            aPtr += 32; bPtr += 32;
        };
        auto storeTile = [&](int buf) {
            *(u32x4*)&lA[buf][row][seg]     = ra0;
            *(u32x4*)&lA[buf][row][seg + 8] = ra1;
            *(u32x4*)&lB[buf][row][seg]     = rb0;
            *(u32x4*)&lB[buf][row][seg + 8] = rb1;
        };
        loadTile();
        storeTile(0);
        for (int it = 0; it < kIters; ++it) {
            const int b = it & 1;
            const bool more = (it + 1) < kIters;
            if (more) loadTile();        // global loads in flight over compute
            __syncthreads();             // buffer b ready; prior reads done
            compute(b);
            if (more) storeTile(1 - b);  // waits loadcnt, fills other buffer
        }
    }

    // C/D layout: VGPR i -> M = i + 8*lhi, N = lrow
    const int cBase = (int)blockIdx.x * 128 + wn + lrow;
    #pragma unroll
    for (int mi = 0; mi < 4; ++mi) {
        #pragma unroll
        for (int ni = 0; ni < 2; ++ni) {
            int col = cBase + ni * 16;
            if (col >= nValid) continue;
            float bAdd = bias ? bias[col] : 0.f;
            #pragma unroll
            for (int i = 0; i < 8; ++i) {
                int r = (int)blockIdx.y * 128 + wm + mi * 16 + lhi * 8 + i;
                if (r < mRows) C[(size_t)r * ldc + col] = acc[mi][ni][i] + bAdd;
            }
        }
    }
}

// ---------------------------------------------------------------------------

extern "C" void kernel_launch(void* const* d_in, const int* in_sizes, int n_in,
                              void* d_out, int out_size, void* d_ws, size_t ws_size,
                              hipStream_t stream) {
    const float* G  = (const float*)d_in[0];
    const float* Wm = (const float*)d_in[1];
    const float* bm = (const float*)d_in[2];
    const float* Wu = (const float*)d_in[3];
    const float* bu = (const float*)d_in[4];
    float* out = (float*)d_out;

    char*  ws  = (char*)d_ws;
    size_t off = 0;
    auto wsAlloc = [&](size_t bytes) -> void* {
        void* p = ws + off;
        off = (off + bytes + 255) & ~(size_t)255;
        return p;
    };
    __bf16* Xbf = (__bf16*)wsAlloc((size_t)NNODES * K2 * sizeof(__bf16)); // 115.2 MB
    float*  C1  = (float*) wsAlloc((size_t)NNODES * K2 * sizeof(float));  // 230.4 MB
    __bf16* W1  = (__bf16*)wsAlloc((size_t)K2 * DP * sizeof(__bf16));     // 8.3 MB
    __bf16* W2  = (__bf16*)wsAlloc((size_t)DP * K2 * sizeof(__bf16));     // 8.3 MB
    float*  m1  = (float*) wsAlloc(2048 * sizeof(float));
    float*  m2  = (float*) wsAlloc(2048 * sizeof(float));
    int*    a1  = (int*)   wsAlloc(2048 * sizeof(int));

    const int TPB = 256;
    const size_t nPad = (size_t)NNODES * DP;

    k_convert_g <<<(unsigned)((nPad + TPB - 1) / TPB), TPB, 0, stream>>>(G, Xbf);
    k_convert_w1<<<(unsigned)(((size_t)K2 * DP + TPB - 1) / TPB), TPB, 0, stream>>>(Wm, W1);
    k_convert_w2<<<(unsigned)(((size_t)DP * K2 + TPB - 1) / TPB), TPB, 0, stream>>>(Wu, W2);

    // GEMM1: [20000 x 1440] x [2880 x 1440]^row -> C1 [20000 x 2880]
    dim3 g1((K2 + 127) / 128, (NNODES + 127) / 128);
    k_wmma_gemm<<<g1, TPB, 0, stream>>>(Xbf, K2, W1, DP, K2,
                                        C1, K2, NNODES, K2, DP / 32, nullptr);

    k_top2<<<(D + TPB - 1) / TPB, TPB, 0, stream>>>(C1, m1, m2, a1);
    k_build_msg<<<(unsigned)((nPad + TPB - 1) / TPB), TPB, 0, stream>>>(C1, bm, m1, m2, a1, Xbf);

    // GEMM2: [20000 x 2880] x [1440 x 2880]^row -> out [20000 x 1433] (+bu)
    dim3 g2((D + 127) / 128, (NNODES + 127) / 128);
    k_wmma_gemm<<<g2, TPB, 0, stream>>>(Xbf, K2, W2, K2, DP,
                                        out, D, NNODES, D, K2 / 32, bu);
}
